// EncoderLayer_32427003085075
// MI455X (gfx1250) — compile-verified, matrix operand order checked by reading, not run on hardware
//
#include <hip/hip_runtime.h>
#include <hip/hip_bf16.h>

typedef __attribute__((ext_vector_type(16))) _Float16 v16h;
typedef __attribute__((ext_vector_type(8)))  _Float16 v8h;
typedef __attribute__((ext_vector_type(4)))  _Float16 v4h;
typedef __attribute__((ext_vector_type(8)))  float    v8f;
typedef __attribute__((ext_vector_type(4)))  float    v4f;

#define D_MODEL 1024
#define HEADS   16
#define DK      64
#define D_FF    4096
#define BATCH   2
#define SEQ     2048
#define ROWS    (BATCH*SEQ)   // 4096

// ---------------------------------------------------------------------------
// fragment helpers (ISA 7.12.2 16-bit A 16x32 layout: lane&15 = row,
// lane>>4 selects K-group offset {0,8}; elements 0..7 -> K=kb+0..7,
// elements 8..15 -> K=kb+16..kb+23)
// ---------------------------------------------------------------------------
__device__ __forceinline__ v16h pack16(v8h lo, v8h hi) {
    v16h r;
#pragma unroll
    for (int i = 0; i < 8; ++i) { r[i] = lo[i]; r[i + 8] = hi[i]; }
    return r;
}
__device__ __forceinline__ v16h load_frag(const _Float16* p) {
    v8h lo = *(const v8h*)(p);
    v8h hi = *(const v8h*)(p + 16);
    return pack16(lo, hi);
}
__device__ __forceinline__ v8f wmma_f16(v16h a, v16h b, v8f c) {
    return __builtin_amdgcn_wmma_f32_16x16x32_f16(false, a, false, b,
                                                  (short)0, c, false, false);
}

// ---------------------------------------------------------------------------
// transpose + f32->f16 convert:  Wt[n*K + k] = W[k*N + n]
// ---------------------------------------------------------------------------
__global__ void k_transpose_cvt(const float* __restrict__ W,
                                _Float16* __restrict__ Wt, int K, int N) {
    int idx = blockIdx.x * 256 + threadIdx.x;
    if (idx >= K * N) return;
    int k = idx / N, n = idx % N;
    Wt[(long)n * K + k] = (_Float16)W[idx];
}

// ---------------------------------------------------------------------------
// LayerNorm (torch Norm semantics: unbiased std, eps added to std) -> f16
// one block (256 threads) per row of D_MODEL=1024
// ---------------------------------------------------------------------------
__global__ void __launch_bounds__(256)
k_layernorm(const float* __restrict__ X, const float* __restrict__ alpha,
            const float* __restrict__ beta, _Float16* __restrict__ Y) {
    __shared__ float s1[256], s2[256];
    const int row = blockIdx.x, t = threadIdx.x;
    const float* xr = X + (long)row * D_MODEL;
    v4f x = *(const v4f*)(xr + t * 4);
    float s = x[0] + x[1] + x[2] + x[3];
    float q = x[0]*x[0] + x[1]*x[1] + x[2]*x[2] + x[3]*x[3];
    s1[t] = s; s2[t] = q; __syncthreads();
    for (int off = 128; off; off >>= 1) {
        if (t < off) { s1[t] += s1[t + off]; s2[t] += s2[t + off]; }
        __syncthreads();
    }
    const float mu  = s1[0] * (1.0f / D_MODEL);
    const float var = (s2[0] - (float)D_MODEL * mu * mu) * (1.0f / (D_MODEL - 1));
    const float inv = 1.0f / (sqrtf(var) + 1e-6f);
    v4h o;
#pragma unroll
    for (int i = 0; i < 4; ++i) {
        int c = t * 4 + i;
        o[i] = (_Float16)(alpha[c] * (x[i] - mu) * inv + beta[c]);
    }
    *(v4h*)(Y + (long)row * D_MODEL + t * 4) = o;
}

// ---------------------------------------------------------------------------
// WMMA GEMM, register-blocked:  C[M,N] = A[M,K](f16) * Bt[N,K](f16)^T + bias
//   - wave tile 32x32 (2x2 accumulators, 4 independent wmma chains per K step,
//     2 A-frags + 2 B-frags -> 16 flop/byte from L2)
//   - block = 256 threads = 8 waves arranged 2(M) x 4(N) -> 64x128 block tile
// OMODE: 0 = f16 plain, 1 = f32 (+resid), 2 = f16 head-scatter [B,H,S,dk],
//        3 = f16 head-transposed [B,H,dk,S]
// ---------------------------------------------------------------------------
template <int OMODE, bool RELU, bool RESID>
__global__ void __launch_bounds__(256)
k_gemm_wmma(const _Float16* __restrict__ A, const _Float16* __restrict__ Bt,
            const float* __restrict__ bias, const float* __restrict__ resid,
            float* __restrict__ Cf, _Float16* __restrict__ Ch,
            int M, int N, int K) {
    const int lane = threadIdx.x & 31;
    const int wave = threadIdx.x >> 5;
    const int wm = wave >> 2, wn = wave & 3;      // 2 x 4 wave grid
    const int bn_cnt = N >> 7;                    // blocks along N (128 wide)
    const int bm = blockIdx.x / bn_cnt;
    const int bn = blockIdx.x % bn_cnt;
    const int m0 = bm * 64  + wm * 32;
    const int n0 = bn * 128 + wn * 32;
    if (m0 >= M) return;                          // uniform per wave

    const int ln = lane & 15;
    const int kb = (lane >> 4) << 3;

    const _Float16* aptr = A  + (long)(m0 + ln) * K + kb;
    const _Float16* bptr = Bt + (long)(n0 + ln) * K + kb;
    const long arow16 = (long)16 * K;

    v8f acc00 = {}, acc01 = {}, acc10 = {}, acc11 = {};
    for (int k = 0; k < K; k += 32) {
        __builtin_prefetch(aptr + 256, 0, 0);     // global_prefetch_b8
        __builtin_prefetch(bptr + 256, 0, 0);
        v16h a0 = load_frag(aptr);
        v16h a1 = load_frag(aptr + arow16);
        v16h b0 = load_frag(bptr);
        v16h b1 = load_frag(bptr + arow16);
        acc00 = wmma_f16(a0, b0, acc00);
        acc01 = wmma_f16(a0, b1, acc01);
        acc10 = wmma_f16(a1, b0, acc10);
        acc11 = wmma_f16(a1, b1, acc11);
        aptr += 32; bptr += 32;
    }

    const int lm = (lane >> 4) << 3;
    v8f accs[2][2] = {{acc00, acc01}, {acc10, acc11}};
#pragma unroll
    for (int i = 0; i < 2; ++i) {
#pragma unroll
        for (int j = 0; j < 2; ++j) {
            const v8f acc = accs[i][j];
            const int col = n0 + j * 16 + ln;
            const float bv = bias[col];
#pragma unroll
            for (int e = 0; e < 8; ++e) {
                const int row = m0 + i * 16 + e + lm;
                float v = acc[e] + bv;
                if (RELU)  v = v > 0.0f ? v : 0.0f;
                if (RESID) v += resid[(long)row * N + col];
                if (OMODE == 0) {
                    Ch[(long)row * N + col] = (_Float16)v;
                } else if (OMODE == 1) {
                    Cf[(long)row * N + col] = v;
                } else if (OMODE == 2) {   // q/k: [B,S,D] -> [B,H,S,dk]
                    int b_ = row / SEQ, s = row % SEQ, h = col >> 6, d = col & 63;
                    Ch[(((long)(b_ * HEADS + h)) * SEQ + s) * DK + d] = (_Float16)v;
                } else {                   // v: [B,S,D] -> [B,H,dk,S]
                    int b_ = row / SEQ, s = row % SEQ, h = col >> 6, d = col & 63;
                    Ch[(((long)(b_ * HEADS + h)) * DK + d) * SEQ + s] = (_Float16)v;
                }
            }
        }
    }
}

// ---------------------------------------------------------------------------
// Fused attention: per block = (b, h, 16-query tile).
// phase1: scores = q.k^T /8 + mask  -> LDS f32 [16][SEQ]   (WMMA)
// phase2: row softmax in LDS
// phase3: ctx = P @ V  via WMMA, A read from LDS, B = V^T (contig in K=S)
// dynamic LDS = 16*SEQ*4 = 128 KB (<320 KB/WGP)
// ---------------------------------------------------------------------------
__global__ void __launch_bounds__(128)
k_attention(const _Float16* __restrict__ QK, const _Float16* __restrict__ VT,
            const int* __restrict__ mask, _Float16* __restrict__ CTX) {
    extern __shared__ float sc[];            // [16][SEQ]
    __shared__ float red[16 * 8];
    __shared__ float rowv[16];

    const int lane = threadIdx.x & 31, wave = threadIdx.x >> 5;
    const int qt = blockIdx.x & 127;
    const int h  = (blockIdx.x >> 7) & 15;
    const int b  = blockIdx.x >> 11;
    const int sq0 = qt << 4;
    const int kb = (lane >> 4) << 3;

    const _Float16* qbase = QK + ((long)(b * HEADS + h)) * SEQ * DK;

    // q fragment (A, 16x64) loaded once
    const _Float16* qp = qbase + (long)(sq0 + (lane & 15)) * DK + kb;
    v16h a_lo = load_frag(qp);
    v16h a_hi = load_frag(qp + 32);

    // ---- phase 1: scores ----
    for (int kt = wave; kt < SEQ / 16; kt += 4) {
        const _Float16* kp = qbase + (long)(kt * 16 + (lane & 15)) * DK + kb;
        v16h b_lo = load_frag(kp);
        v16h b_hi = load_frag(kp + 32);
        v8f s = {};
        s = wmma_f16(a_lo, b_lo, s);
        s = wmma_f16(a_hi, b_hi, s);
        const int key = kt * 16 + (lane & 15);
        const int mk  = mask[b * SEQ + key];
#pragma unroll
        for (int e = 0; e < 8; ++e) {
            const int qrow = e + ((lane >> 4) << 3);
            sc[qrow * SEQ + key] = (mk == 0) ? -1e9f : s[e] * 0.125f;
        }
    }
    __syncthreads();

    // ---- phase 2: softmax (8 threads per row) ----
    const int r = threadIdx.x >> 3, sub = threadIdx.x & 7;
    float m = -3.4e38f;
    for (int j = sub; j < SEQ; j += 8) m = fmaxf(m, sc[r * SEQ + j]);
    red[r * 8 + sub] = m; __syncthreads();
    if (sub == 0) {
        float mm = red[r * 8];
        for (int i = 1; i < 8; ++i) mm = fmaxf(mm, red[r * 8 + i]);
        rowv[r] = mm;
    }
    __syncthreads();
    const float rm = rowv[r];
    float sum = 0.0f;
    for (int j = sub; j < SEQ; j += 8) {
        float e = __expf(sc[r * SEQ + j] - rm);
        sc[r * SEQ + j] = e; sum += e;
    }
    red[r * 8 + sub] = sum; __syncthreads();
    if (sub == 0) {
        float ss = 0.0f;
        for (int i = 0; i < 8; ++i) ss += red[r * 8 + i];
        rowv[r] = 1.0f / ss;
    }
    __syncthreads();
    const float inv = rowv[r];
    for (int j = sub; j < SEQ; j += 8) sc[r * SEQ + j] *= inv;
    __syncthreads();

    // ---- phase 3: ctx = P @ V, each wave owns one 16-wide slice of dk ----
    const int n0 = wave << 4;
    const _Float16* vbase = VT + ((long)(b * HEADS + h)) * DK * SEQ;
    const float*    ap = sc + (lane & 15) * SEQ + kb;
    const _Float16* vp = vbase + (long)(n0 + (lane & 15)) * SEQ + kb;

    v8f acc = {};
    for (int k = 0; k < SEQ; k += 32) {
        v4f f0 = *(const v4f*)(ap + k);
        v4f f1 = *(const v4f*)(ap + k + 4);
        v4f f2 = *(const v4f*)(ap + k + 16);
        v4f f3 = *(const v4f*)(ap + k + 20);
        v16h a;
#pragma unroll
        for (int i = 0; i < 4; ++i) {
            a[i]      = (_Float16)f0[i];
            a[4 + i]  = (_Float16)f1[i];
            a[8 + i]  = (_Float16)f2[i];
            a[12 + i] = (_Float16)f3[i];
        }
        v16h bf = load_frag(vp + k);
        acc = wmma_f16(a, bf, acc);
    }
#pragma unroll
    for (int e = 0; e < 8; ++e) {
        const int qrow = e + ((lane >> 4) << 3);
        CTX[((long)(b * SEQ + sq0 + qrow)) * D_MODEL + h * DK + n0 + (lane & 15)] =
            (_Float16)acc[e];
    }
}

// ---------------------------------------------------------------------------
// host launcher
// ---------------------------------------------------------------------------
extern "C" void kernel_launch(void* const* d_in, const int* in_sizes, int n_in,
                              void* d_out, int out_size, void* d_ws, size_t ws_size,
                              hipStream_t stream) {
    const float* x      = (const float*)d_in[0];
    const int*   mask   = (const int*)  d_in[1];
    const float* alpha1 = (const float*)d_in[2];
    const float* bias1  = (const float*)d_in[3];
    const float* alpha2 = (const float*)d_in[4];
    const float* bias2  = (const float*)d_in[5];
    const float* Wk     = (const float*)d_in[6];
    const float* bk     = (const float*)d_in[7];
    const float* Wv     = (const float*)d_in[8];
    const float* bv     = (const float*)d_in[9];
    const float* Wo     = (const float*)d_in[10];
    const float* bo     = (const float*)d_in[11];
    const float* W1     = (const float*)d_in[12];
    const float* b1     = (const float*)d_in[13];
    const float* W2     = (const float*)d_in[14];
    const float* b2     = (const float*)d_in[15];
    float* out = (float*)d_out;

    // workspace layout (bytes), ~110 MB total
    char* ws = (char*)d_ws;
    size_t off = 0;
    _Float16* x2h  = (_Float16*)(ws + off); off += (size_t)ROWS * D_MODEL * 2;
    _Float16* x2bh = (_Float16*)(ws + off); off += (size_t)ROWS * D_MODEL * 2;
    _Float16* qkh  = (_Float16*)(ws + off); off += (size_t)BATCH*HEADS*SEQ*DK*2;
    _Float16* vth  = (_Float16*)(ws + off); off += (size_t)BATCH*HEADS*DK*SEQ*2;
    _Float16* ctxh = (_Float16*)(ws + off); off += (size_t)ROWS * D_MODEL * 2;
    _Float16* h1h  = (_Float16*)(ws + off); off += (size_t)ROWS * D_FF * 2;
    _Float16* Wkt  = (_Float16*)(ws + off); off += (size_t)D_MODEL * D_MODEL * 2;
    _Float16* Wvt  = (_Float16*)(ws + off); off += (size_t)D_MODEL * D_MODEL * 2;
    _Float16* Wot  = (_Float16*)(ws + off); off += (size_t)D_MODEL * D_MODEL * 2;
    _Float16* W1t  = (_Float16*)(ws + off); off += (size_t)D_MODEL * D_FF * 2;
    _Float16* W2t  = (_Float16*)(ws + off); off += (size_t)D_FF * D_MODEL * 2;
    float*    x1   = (float*)   (ws + off); off += (size_t)ROWS * D_MODEL * 4;

    // 1) weight transpose + cvt (B^T layouts, contiguous in K)
    k_transpose_cvt<<<(D_MODEL*D_MODEL + 255)/256, 256, 0, stream>>>(Wk, Wkt, D_MODEL, D_MODEL);
    k_transpose_cvt<<<(D_MODEL*D_MODEL + 255)/256, 256, 0, stream>>>(Wv, Wvt, D_MODEL, D_MODEL);
    k_transpose_cvt<<<(D_MODEL*D_MODEL + 255)/256, 256, 0, stream>>>(Wo, Wot, D_MODEL, D_MODEL);
    k_transpose_cvt<<<(D_MODEL*D_FF    + 255)/256, 256, 0, stream>>>(W1, W1t, D_MODEL, D_FF);
    k_transpose_cvt<<<(D_FF*D_MODEL    + 255)/256, 256, 0, stream>>>(W2, W2t, D_FF, D_MODEL);

    // 2) LN1 -> f16
    k_layernorm<<<ROWS, 256, 0, stream>>>(x, alpha1, bias1, x2h);

    // GEMM grids: block tile 64(M) x 128(N), 256 threads
    #define GEMM_GRID(M_, N_) (((M_) / 64) * ((N_) / 128))

    // 3) q (== k, source bug preserved): x2 @ Wk + bk -> [B,H,S,dk]
    k_gemm_wmma<2, false, false><<<GEMM_GRID(ROWS, D_MODEL), 256, 0, stream>>>(
        x2h, Wkt, bk, nullptr, nullptr, qkh, ROWS, D_MODEL, D_MODEL);
    // 4) v: x2 @ Wv + bv -> [B,H,dk,S]  (transposed for P@V K-contiguity)
    k_gemm_wmma<3, false, false><<<GEMM_GRID(ROWS, D_MODEL), 256, 0, stream>>>(
        x2h, Wvt, bv, nullptr, nullptr, vth, ROWS, D_MODEL, D_MODEL);

    // 5) fused attention -> ctx (concat layout), 128 KB dynamic LDS
    k_attention<<<BATCH*HEADS*(SEQ/16), 128, 16*SEQ*sizeof(float), stream>>>(
        qkh, vth, mask, ctxh);

    // 6) x1 = x + ctx @ Wo + bo   (f32)
    k_gemm_wmma<1, false, true><<<GEMM_GRID(ROWS, D_MODEL), 256, 0, stream>>>(
        ctxh, Wot, bo, x, x1, nullptr, ROWS, D_MODEL, D_MODEL);

    // 7) LN2 -> f16
    k_layernorm<<<ROWS, 256, 0, stream>>>(x1, alpha2, bias2, x2bh);

    // 8) h1 = relu(x2b @ W1 + b1)  (f16)
    k_gemm_wmma<0, true, false><<<GEMM_GRID(ROWS, D_FF), 256, 0, stream>>>(
        x2bh, W1t, b1, nullptr, nullptr, h1h, ROWS, D_FF, D_MODEL);

    // 9) out = x1 + h1 @ W2 + b2  (f32 -> d_out)
    k_gemm_wmma<1, false, true><<<GEMM_GRID(ROWS, D_MODEL), 256, 0, stream>>>(
        h1h, W2t, b2, x1, out, nullptr, ROWS, D_MODEL, D_FF);

    (void)in_sizes; (void)n_in; (void)out_size; (void)ws_size;
}
